// VectorQuantizer_37134287241731
// MI455X (gfx1250) — matrix operand order repaired and checked by reading, not compiled, not run on hardware
//
#include <hip/hip_runtime.h>
#include <hip/hip_bf16.h>

typedef __attribute__((ext_vector_type(16))) __bf16 v16bf;
typedef __attribute__((ext_vector_type(8)))  __bf16 bf16x8;
typedef __attribute__((ext_vector_type(8)))  float  v8f;

#define K_CODES 8192
#define DIM     256
#define NTOK    32768      // 16 * 2048
#define LDS_ROW 264        // 256 + 8 bf16 pad (breaks 512B bank stride)

// float -> bf16, round to nearest even (avoids relying on __bf16 cvt support)
__device__ __forceinline__ __bf16 f2bf(float f) {
    unsigned u = __float_as_uint(f);
    unsigned r = (u + 0x7fffu + ((u >> 16) & 1u)) >> 16;
    unsigned short us = (unsigned short)r;
    return __builtin_bit_cast(__bf16, us);
}

__device__ __forceinline__ void merge_min(float& bv, int& bi, float ov, int oi) {
    if (ov < bv || (ov == bv && oi < bi)) { bv = ov; bi = oi; }
}

// ---------------- Kernel A: codebook -> bf16 + squared norms ----------------
__global__ void vq_prep(const float* __restrict__ cb,
                        __bf16* __restrict__ cb_bf,
                        float* __restrict__ c2) {
    const int k = blockIdx.x;
    const int d = threadIdx.x;
    float v = cb[(size_t)k * DIM + d];
    cb_bf[(size_t)k * DIM + d] = f2bf(v);
    __shared__ float red[256];
    red[d] = v * v;
    __syncthreads();
    for (int s = 128; s > 0; s >>= 1) {
        if (d < s) red[d] += red[d + s];
        __syncthreads();
    }
    if (d == 0) c2[k] = red[0];
}

// ---------------- Kernel B: fused bf16-WMMA distance + argmin ----------------
__launch_bounds__(256, 1)
__global__ void vq_argmin(const float* __restrict__ x,
                          const __bf16* __restrict__ cb_bf,
                          const float* __restrict__ c2,
                          int* __restrict__ idx_out,
                          float* __restrict__ idx_out_f) {
    const int lane = threadIdx.x & 31;
    const int wave = threadIdx.x >> 5;
    const int m0   = blockIdx.x * 256 + wave * 32;   // 32 tokens per wave
    const int mrow = lane & 15;                       // A: M row / B,C: N column
    const int kh   = lane >> 4;                       // K-half selector

    // ---- Preload A: 2 tiles of 16 tokens x 256 dims, bf16 WMMA layout ----
    // 16-bit A 16x32: lanes 0-15 hold K 0-7 (elems 0-7) and K 16-23 (elems 8-15);
    // lanes 16-31 hold K 8-15 and 24-31.  => elems 0-7 at d+kh*8, 8-15 at d+16+kh*8.
    v16bf a[2][8];
    #pragma unroll
    for (int t = 0; t < 2; ++t) {
        const float* xr = x + (size_t)(m0 + 16 * t + mrow) * DIM;
        #pragma unroll
        for (int c = 0; c < 8; ++c) {
            const float* p0 = xr + c * 32 + kh * 8;
            const float* p1 = xr + c * 32 + 16 + kh * 8;
            float4 f0 = *(const float4*)(p0);
            float4 f1 = *(const float4*)(p0 + 4);
            float4 f2 = *(const float4*)(p1);
            float4 f3 = *(const float4*)(p1 + 4);
            v16bf av;
            av[0]  = f2bf(f0.x); av[1]  = f2bf(f0.y); av[2]  = f2bf(f0.z); av[3]  = f2bf(f0.w);
            av[4]  = f2bf(f1.x); av[5]  = f2bf(f1.y); av[6]  = f2bf(f1.z); av[7]  = f2bf(f1.w);
            av[8]  = f2bf(f2.x); av[9]  = f2bf(f2.y); av[10] = f2bf(f2.z); av[11] = f2bf(f2.w);
            av[12] = f2bf(f3.x); av[13] = f2bf(f3.y); av[14] = f2bf(f3.z); av[15] = f2bf(f3.w);
            a[t][c] = av;
        }
    }

    __shared__ __bf16 ldsB[2][16 * LDS_ROW];   // double-buffered 16-code panel

    const int scode = threadIdx.x >> 4;   // 0..15 : code within tile
    const int sseg  = threadIdx.x & 15;   // 0..15 : 16-elem segment of the row
    auto stage = [&](int kt, int buf) {
        const uint4* src =
            (const uint4*)(cb_bf + (size_t)(kt * 16 + scode) * DIM + sseg * 16);
        uint4* dst = (uint4*)(&ldsB[buf][scode * LDS_ROW + sseg * 16]);
        dst[0] = src[0];
        dst[1] = src[1];
    };

    float best0[8], best1[8];
    int   bid0[8],  bid1[8];
    #pragma unroll
    for (int j = 0; j < 8; ++j) {
        best0[j] = best1[j] = 3.4e38f;
        bid0[j]  = bid1[j]  = 0;
    }

    stage(0, 0);
    const int NT = K_CODES / 16;   // 512 code tiles
    for (int kt = 0; kt < NT; ++kt) {
        __syncthreads();                       // panel kt ready
        if (kt + 1 < NT) stage(kt + 1, (kt + 1) & 1);

        const __bf16* bbuf = ldsB[kt & 1];
        v8f acc0 = {};
        v8f acc1 = {};
        #pragma unroll
        for (int c = 0; c < 8; ++c) {
            // B 32x16 layout: lane = column mrow; elems 0-7 at K kh*8, 8-15 at K 16+kh*8.
            const __bf16* bp = bbuf + mrow * LDS_ROW + c * 32 + kh * 8;
            bf16x8 lo = *(const bf16x8*)(bp);
            bf16x8 hi = *(const bf16x8*)(bp + 16);
            v16bf bv = __builtin_shufflevector(lo, hi, 0, 1, 2, 3, 4, 5, 6, 7,
                                               8, 9, 10, 11, 12, 13, 14, 15);
            acc0 = __builtin_amdgcn_wmma_f32_16x16x32_bf16(
                false, a[0][c], false, bv, (short)0, acc0, false, false);
            acc1 = __builtin_amdgcn_wmma_f32_16x16x32_bf16(
                false, a[1][c], false, bv, (short)0, acc1, false, false);
        }

        const int   code = kt * 16 + mrow;
        const float c2v  = c2[code];
        #pragma unroll
        for (int j = 0; j < 8; ++j) {
            float d0 = __builtin_fmaf(-2.0f, acc0[j], c2v);
            float d1 = __builtin_fmaf(-2.0f, acc1[j], c2v);
            if (d0 < best0[j]) { best0[j] = d0; bid0[j] = code; }
            if (d1 < best1[j]) { best1[j] = d1; bid1[j] = code; }
        }
    }

    // ---- argmin across the 16 columns of each half-wave (ds_swizzle xor) ----
    #pragma unroll
    for (int t = 0; t < 2; ++t) {
        #pragma unroll
        for (int j = 0; j < 8; ++j) {
            float bv = (t == 0) ? best0[j] : best1[j];
            int   bi = (t == 0) ? bid0[j]  : bid1[j];

            float ov; int oi;
            ov = __int_as_float(__builtin_amdgcn_ds_swizzle(__float_as_int(bv), 0x041f));
            oi = __builtin_amdgcn_ds_swizzle(bi, 0x041f);
            merge_min(bv, bi, ov, oi);
            ov = __int_as_float(__builtin_amdgcn_ds_swizzle(__float_as_int(bv), 0x081f));
            oi = __builtin_amdgcn_ds_swizzle(bi, 0x081f);
            merge_min(bv, bi, ov, oi);
            ov = __int_as_float(__builtin_amdgcn_ds_swizzle(__float_as_int(bv), 0x101f));
            oi = __builtin_amdgcn_ds_swizzle(bi, 0x101f);
            merge_min(bv, bi, ov, oi);
            ov = __int_as_float(__builtin_amdgcn_ds_swizzle(__float_as_int(bv), 0x201f));
            oi = __builtin_amdgcn_ds_swizzle(bi, 0x201f);
            merge_min(bv, bi, ov, oi);

            // C layout: lanes 0-15 hold M=j, lanes 16-31 hold M=8+j
            if (lane == 0) {
                int tok = m0 + 16 * t + j;
                idx_out[tok]   = bi;
                idx_out_f[tok] = (float)bi;
            }
            if (lane == 16) {
                int tok = m0 + 16 * t + 8 + j;
                idx_out[tok]   = bi;
                idx_out_f[tok] = (float)bi;
            }
        }
    }
}

// ---------------- Kernel C: gather + per-block loss partials ----------------
__global__ void vq_gather(const float* __restrict__ x,
                          const float* __restrict__ cb,
                          const int* __restrict__ idx,
                          float* __restrict__ out_q,
                          float* __restrict__ partials) {
    const int e4 = blockIdx.x * 256 + threadIdx.x;  // quad index
    const int n  = e4 >> 6;                         // token (64 quads / token)
    const int d  = (e4 & 63) << 2;                  // dim
    const int k  = idx[n];
    float4 q  = *(const float4*)(cb + (size_t)k * DIM + d);
    float4 xv = *(const float4*)(x  + (size_t)n * DIM + d);
    *(float4*)(out_q + (size_t)n * DIM + d) = q;
    float dx = q.x - xv.x, dy = q.y - xv.y, dz = q.z - xv.z, dw = q.w - xv.w;
    float s = dx * dx + dy * dy + dz * dz + dw * dw;

    __shared__ float red[256];
    red[threadIdx.x] = s;
    __syncthreads();
    for (int st = 128; st > 0; st >>= 1) {
        if (threadIdx.x < st) red[threadIdx.x] += red[threadIdx.x + st];
        __syncthreads();
    }
    if (threadIdx.x == 0) partials[blockIdx.x] = red[0];
}

// ---------------- Kernel D: deterministic final loss reduction ----------------
__global__ void vq_loss(const float* __restrict__ partials,
                        float* __restrict__ out_loss) {
    __shared__ float red[256];
    float s = 0.0f;
    for (int i = threadIdx.x; i < 8192; i += 256) s += partials[i];
    red[threadIdx.x] = s;
    __syncthreads();
    for (int st = 128; st > 0; st >>= 1) {
        if (threadIdx.x < st) red[threadIdx.x] += red[threadIdx.x + st];
        __syncthreads();
    }
    // loss = q_latent + 0.25*e_latent ; forward values identical => 1.25 * MSE
    if (threadIdx.x == 0) out_loss[0] = 1.25f * red[0] / 8388608.0f;
}

extern "C" void kernel_launch(void* const* d_in, const int* in_sizes, int n_in,
                              void* d_out, int out_size, void* d_ws, size_t ws_size,
                              hipStream_t stream) {
    const float* x  = (const float*)d_in[0];   // [16,2048,256] f32
    const float* cb = (const float*)d_in[1];   // [8192,256]    f32

    char* ws = (char*)d_ws;
    __bf16* cb_bf   = (__bf16*)(ws);                                   // 4 MiB
    float*  c2      = (float*)(ws + 4u * 1024 * 1024);                 // 32 KiB
    int*    idx     = (int*)  (ws + 4u * 1024 * 1024 + 64 * 1024);     // 128 KiB
    float*  partial = (float*)(ws + 4u * 1024 * 1024 + 64 * 1024 + 256 * 1024); // 32 KiB

    float* out_q    = (float*)d_out;                 // [N*D] quantized (== q, STE)
    float* out_loss = out_q + (size_t)NTOK * DIM;    // [1]
    float* out_idx  = out_loss + 1;                  // [N] indices as float

    vq_prep  <<<K_CODES, 256, 0, stream>>>(cb, cb_bf, c2);
    vq_argmin<<<NTOK / 256, 256, 0, stream>>>(x, cb_bf, c2, idx, out_idx);
    vq_gather<<<(NTOK * DIM / 4) / 256, 256, 0, stream>>>(x, cb, idx, out_q, partial);
    vq_loss  <<<1, 256, 0, stream>>>(partial, out_loss);
}
